// Quantizer_19928648253941
// MI455X (gfx1250) — matrix operand order repaired and checked by reading, not compiled, not run on hardware
//
#include <hip/hip_runtime.h>

// ---- types -----------------------------------------------------------------
typedef __attribute__((ext_vector_type(16))) __bf16 v16bf;
typedef __attribute__((ext_vector_type(8)))  __bf16 v8bf;
typedef __attribute__((ext_vector_type(4)))  __bf16 v4bf;
typedef __attribute__((ext_vector_type(8)))  float  v8f;
typedef __attribute__((ext_vector_type(4)))  float  v4f;

#define DIM            512
#define CODEBOOKS      16
#define CODEBOOK_SIZE  256
#define N_OUT          (CODEBOOKS * CODEBOOK_SIZE)   // 4096
#define BATCH          32768
#define ROWS_PER_BLOCK 32        // 2 M-tiles of 16
#define SCALE_SPEED    10.0f

// ============================================================================
// Pre-pass: split W (f32) into bf16 hi/lo once. W is reused by all 1024 row
// blocks of the encoder, so paying the f32->bf16x2 split once (8 MB read,
// 8 MB write, ~0.7us at HBM speed) removes ~20 VALU cvt ops from every
// iteration of the encoder's hot K-loop.
// ============================================================================
__global__ __launch_bounds__(256)
void wsplit_kernel(const float* __restrict__ W,
                   __bf16*      __restrict__ Whi,
                   __bf16*      __restrict__ Wlo)
{
    const int t = blockIdx.x * 256 + threadIdx.x;    // one float4 per thread
    v4f v = ((const v4f*)W)[t];
    v4bf h, l;
#pragma unroll
    for (int e = 0; e < 4; ++e) {
        __bf16 hh = (__bf16)v[e];
        h[e] = hh;
        l[e] = (__bf16)(v[e] - (float)hh);
    }
    ((v4bf*)Whi)[t] = h;
    ((v4bf*)Wlo)[t] = l;
}

// ============================================================================
// Encode: fused GEMM(x @ W^T + b) + per-codebook argmax.
// Grid: (BATCH/32) blocks x 512 threads (16 waves). Wave w handles codebook w
// for the block's 32 rows. Split-bf16 (hi+lo) -> 3x v_wmma_f32_16x16x32_bf16
// gives ~f32-grade logits at bf16-WMMA throughput.
// logits_scale is argmax-invariant (exp > 0) and therefore skipped.
// ============================================================================
__global__ __launch_bounds__(512)
void encode_kernel(const float*  __restrict__ x,
                   const __bf16* __restrict__ Whi,
                   const __bf16* __restrict__ Wlo,
                   const float*  __restrict__ bias,
                   int*          __restrict__ idx_out)
{
    __shared__ __bf16 sAhi[ROWS_PER_BLOCK * DIM];   // 32 KB
    __shared__ __bf16 sAlo[ROWS_PER_BLOCK * DIM];   // 32 KB

    const int tid  = threadIdx.x;
    const int lane = tid & 31;
    const int cb   = tid >> 5;              // wave id == codebook id (0..15)
    const int m0   = blockIdx.x * ROWS_PER_BLOCK;

    // ---- stage x tile to LDS as (hi, lo) bf16 pair ------------------------
    const v4f* xq = (const v4f*)x;
    for (int t = tid; t < ROWS_PER_BLOCK * (DIM / 4); t += 512) {
        const int row = t >> 7;             // DIM/4 = 128 float4 per row
        const int kq  = t & 127;
        v4f v = xq[(size_t)(m0 + row) * (DIM / 4) + kq];
        v4bf h, l;
#pragma unroll
        for (int e = 0; e < 4; ++e) {
            __bf16 hh = (__bf16)v[e];
            h[e] = hh;
            l[e] = (__bf16)(v[e] - (float)hh);
        }
        *(v4bf*)&sAhi[row * DIM + kq * 4] = h;
        *(v4bf*)&sAlo[row * DIM + kq * 4] = l;
    }
    __syncthreads();

    const int nsub = lane & 15;             // C/D col, B's N lane
    const int half = lane >> 4;             // K-half selector

    float bestv[2][8];
    int   besti[2][8];
#pragma unroll
    for (int mt = 0; mt < 2; ++mt)
#pragma unroll
        for (int i = 0; i < 8; ++i) { bestv[mt][i] = -3.402823466e38f; besti[mt][i] = 0; }

#pragma unroll 1
    for (int nt = 0; nt < 16; ++nt) {       // 16 N-tiles = one 256-wide codebook
        v8f acc[2];
#pragma unroll
        for (int mt = 0; mt < 2; ++mt)
#pragma unroll
            for (int i = 0; i < 8; ++i) acc[mt][i] = 0.0f;

#pragma unroll 1
        for (int kt = 0; kt < 16; ++kt) {   // K = 512 in steps of 32
            // ---- B fragments: 16 contiguous pre-split bf16 per lane -------
            const size_t woff = (size_t)(cb * CODEBOOK_SIZE + nt * 16 + nsub) * DIM
                              + kt * 32 + half * 16;
            v8bf bh0 = *(const v8bf*)(Whi + woff);
            v8bf bh1 = *(const v8bf*)(Whi + woff + 8);
            v8bf bl0 = *(const v8bf*)(Wlo + woff);
            v8bf bl1 = *(const v8bf*)(Wlo + woff + 8);
            v16bf bhi = __builtin_shufflevector(bh0, bh1, 0,1,2,3,4,5,6,7,8,9,10,11,12,13,14,15);
            v16bf blo = __builtin_shufflevector(bl0, bl1, 0,1,2,3,4,5,6,7,8,9,10,11,12,13,14,15);

            // ---- A fragments from LDS, 2 M-tiles, 2x16B loads each --------
#pragma unroll
            for (int mt = 0; mt < 2; ++mt) {
                const int off = (mt * 16 + nsub) * DIM + kt * 32 + half * 16;
                v8bf h0 = *(const v8bf*)&sAhi[off];
                v8bf h1 = *(const v8bf*)&sAhi[off + 8];
                v8bf l0 = *(const v8bf*)&sAlo[off];
                v8bf l1 = *(const v8bf*)&sAlo[off + 8];
                v16bf ahi = __builtin_shufflevector(h0, h1, 0,1,2,3,4,5,6,7,8,9,10,11,12,13,14,15);
                v16bf alo = __builtin_shufflevector(l0, l1, 0,1,2,3,4,5,6,7,8,9,10,11,12,13,14,15);
                acc[mt] = __builtin_amdgcn_wmma_f32_16x16x32_bf16(false, ahi, false, bhi,
                                                                  (short)0, acc[mt], false, false);
                acc[mt] = __builtin_amdgcn_wmma_f32_16x16x32_bf16(false, ahi, false, blo,
                                                                  (short)0, acc[mt], false, false);
                acc[mt] = __builtin_amdgcn_wmma_f32_16x16x32_bf16(false, alo, false, bhi,
                                                                  (short)0, acc[mt], false, false);
            }
        }

        // ---- bias + running per-lane argmax (col = nsub for all 8 regs) ---
        const float bv   = bias[cb * CODEBOOK_SIZE + nt * 16 + nsub];
        const int   nidx = nt * 16 + nsub;
#pragma unroll
        for (int mt = 0; mt < 2; ++mt)
#pragma unroll
            for (int i = 0; i < 8; ++i) {
                float val = acc[mt][i] + bv;
                if (val > bestv[mt][i]) { bestv[mt][i] = val; besti[mt][i] = nidx; }
            }
    }

    // ---- cross-lane argmax within each 16-lane half, then store -----------
#pragma unroll
    for (int mt = 0; mt < 2; ++mt)
#pragma unroll
        for (int i = 0; i < 8; ++i) {
            float v  = bestv[mt][i];
            int   ix = besti[mt][i];
#pragma unroll
            for (int m = 8; m >= 1; m >>= 1) {
                float ov = __shfl_xor(v,  m, 32);
                int   oi = __shfl_xor(ix, m, 32);
                if (ov > v || (ov == v && oi < ix)) { v = ov; ix = oi; }
            }
            if (nsub == 0) {
                const int row = m0 + mt * 16 + i + half * 8;   // C-layout row
                idx_out[row * CODEBOOKS + cb] = ix;
            }
        }
}

// ============================================================================
// Decode: out[row] = exp(10*centers_scale) * sum_cb centers[cb, idx[row,cb], :]
// One block per row, 128 threads x float4 = 512 dims. centers (8 MB) is
// L2-resident, so the 16-way gather is an L2-bandwidth problem.
// ============================================================================
__global__ __launch_bounds__(128)
void decode_kernel(const float* __restrict__ centers,
                   const float* __restrict__ centers_scale,
                   const int*   __restrict__ idx,
                   float*       __restrict__ out)
{
    const int row = blockIdx.x;
    const int dq  = threadIdx.x;                  // 0..127 float4 slots
    const float s = __expf(centers_scale[0] * SCALE_SPEED);

    const v4f* cq = (const v4f*)centers;
    v4f sum = {0.0f, 0.0f, 0.0f, 0.0f};
#pragma unroll
    for (int cb = 0; cb < CODEBOOKS; ++cb) {
        const int code = idx[row * CODEBOOKS + cb];
        sum += cq[(size_t)(cb * CODEBOOK_SIZE + code) * (DIM / 4) + dq];
    }
    ((v4f*)out)[(size_t)row * (DIM / 4) + dq] = sum * s;
}

// ============================================================================
extern "C" void kernel_launch(void* const* d_in, const int* in_sizes, int n_in,
                              void* d_out, int out_size, void* d_ws, size_t ws_size,
                              hipStream_t stream)
{
    const float* x             = (const float*)d_in[0];   // (32768, 512)
    const float* W             = (const float*)d_in[1];   // (4096, 512)
    const float* b             = (const float*)d_in[2];   // (4096,)
    const float* centers       = (const float*)d_in[3];   // (16, 256, 512)
    // d_in[4] = logits_scale: argmax-invariant, intentionally unused.
    const float* centers_scale = (const float*)d_in[5];   // scalar

    // Workspace layout: [0,2MB) indexes | [2MB,6MB) W_hi | [6MB,10MB) W_lo
    int*    idx_buf = (int*)d_ws;
    __bf16* Whi     = (__bf16*)((char*)d_ws + (2u << 20));
    __bf16* Wlo     = (__bf16*)((char*)d_ws + (6u << 20));

    wsplit_kernel<<<(N_OUT * DIM / 4) / 256, 256, 0, stream>>>(W, Whi, Wlo);
    encode_kernel<<<BATCH / ROWS_PER_BLOCK, 512, 0, stream>>>(x, Whi, Wlo, b, idx_buf);
    decode_kernel<<<BATCH, 128, 0, stream>>>(centers, centers_scale, idx_buf,
                                             (float*)d_out);
}
